// ScaledDotAttention_10711648436813
// MI455X (gfx1250) — matrix operand order repaired and checked by reading, not compiled, not run on hardware
//
#include <hip/hip_runtime.h>
#include <hip/hip_bf16.h>

#define BATCH 8
#define NTOK  2048
#define DDIM  512

typedef __bf16          bf16x16 __attribute__((ext_vector_type(16)));
typedef float           f32x8   __attribute__((ext_vector_type(8)));
typedef unsigned short  u16x16  __attribute__((ext_vector_type(16)));
typedef unsigned short  u16x8   __attribute__((ext_vector_type(8)));

struct pair16 { uint4 lo, hi; };   // 32 bytes, 16B-aligned loads

// hardware bf16 convert (RNE) via clang fptrunc
static __device__ __forceinline__ unsigned short bfbits(float f) {
    __bf16 h = (__bf16)f;
    return __builtin_bit_cast(unsigned short, h);
}

// load 32 contiguous bytes as a bf16x16 fragment (16B alignment required)
static __device__ __forceinline__ u16x16 load32B(const unsigned short* p) {
    pair16 t;
    t.lo = *(const uint4*)p;
    t.hi = *(const uint4*)(p + 8);
    return __builtin_bit_cast(u16x16, t);
}
// load two 16B runs at p and p+16 elements (A-fragment pattern)
static __device__ __forceinline__ u16x16 load2x16(const unsigned short* p) {
    pair16 t;
    t.lo = *(const uint4*)p;
    t.hi = *(const uint4*)(p + 16);
    return __builtin_bit_cast(u16x16, t);
}

static __device__ __forceinline__ f32x8 wmma_bb(u16x16 a, u16x16 b, f32x8 c) {
    return __builtin_amdgcn_wmma_f32_16x16x32_bf16(
        false, __builtin_bit_cast(bf16x16, a),
        false, __builtin_bit_cast(bf16x16, b),
        (short)0, c, false, false);
}

// ---------------------------------------------------------------------------
// Projection: Out(bf16) = (X(f32)[16384x512] @ W(f32)[512x512]) * oscale
// transpose==0 : Out row-major [B*N][D]
// transpose==1 : Out = [B][D][N]  (V stored transposed for attn@V GEMM)
// ---------------------------------------------------------------------------
#define WT_STRIDE 40  // padded bf16 row stride: 80B = 16B-aligned, bank-spread

__global__ __launch_bounds__(256) void proj_kernel(
    const float* __restrict__ X, const float* __restrict__ W,
    unsigned short* __restrict__ Out, float oscale, int transpose)
{
    __shared__ unsigned short WT[128 * WT_STRIDE];  // W tile, transposed, bf16

    const int m0   = blockIdx.x * 128;
    const int n0   = blockIdx.y * 128;
    const int t    = threadIdx.x;
    const int wave = t >> 5, lane = t & 31;
    const int half = lane >> 4, l16 = lane & 15;
    const int mrow = m0 + wave * 16 + l16;   // A-fragment row for this lane

    f32x8 acc[8];
    f32x8 zero = {};
#pragma unroll
    for (int i = 0; i < 8; ++i) acc[i] = zero;

    for (int kt = 0; kt < 16; ++kt) {
        const int k0 = kt * 32;
        // stage W[k0:k0+32][n0:n0+128] transposed -> WT[nn][kk] (bf16), float4 loads
#pragma unroll
        for (int j = 0; j < 4; ++j) {
            int q4 = t + 256 * j;            // float4 index 0..1023
            int kk = q4 >> 5;                // 0..31
            int nn = (q4 & 31) * 4;          // 0..124
            float4 w = *(const float4*)&W[(size_t)(k0 + kk) * DDIM + (n0 + nn)];
            WT[(nn + 0) * WT_STRIDE + kk] = bfbits(w.x);
            WT[(nn + 1) * WT_STRIDE + kk] = bfbits(w.y);
            WT[(nn + 2) * WT_STRIDE + kk] = bfbits(w.z);
            WT[(nn + 3) * WT_STRIDE + kk] = bfbits(w.w);
        }
        __syncthreads();

        // A fragment: element e <-> K = (e>>3)*16 + 8*half + (e&7), f32 -> bf16
        bf16x16 av;
        {
            const float* xp = X + (size_t)mrow * DDIM + k0 + half * 8;
#pragma unroll
            for (int e = 0; e < 8; ++e) av[e]     = (__bf16)xp[e];
#pragma unroll
            for (int e = 0; e < 8; ++e) av[8 + e] = (__bf16)xp[16 + e];
        }
        u16x16 au = __builtin_bit_cast(u16x16, av);

        // preload all 8 B fragments, then 8 WMMAs (graduated dscnt waits)
        u16x16 bu[8];
#pragma unroll
        for (int ct = 0; ct < 8; ++ct)
            bu[ct] = load2x16(&WT[(ct * 16 + l16) * WT_STRIDE + half * 8]);
#pragma unroll
        for (int ct = 0; ct < 8; ++ct)
            acc[ct] = wmma_bb(au, bu[ct], acc[ct]);
        __syncthreads();
    }

    if (!transpose) {
#pragma unroll
        for (int ct = 0; ct < 8; ++ct) {
            int n = n0 + ct * 16 + l16;
#pragma unroll
            for (int r = 0; r < 8; ++r) {   // C elem r -> row r + 8*half
                int m = m0 + wave * 16 + r + half * 8;
                Out[(size_t)m * DDIM + n] = bfbits(acc[ct][r] * oscale);
            }
        }
    } else {
        // V transposed: VT[b][d][token]; lane's 8 tokens contiguous -> 16B store
        const int mbase = m0 + wave * 16 + half * 8;
        const int b     = mbase >> 11;          // tile never straddles batches
        const int tok   = mbase & (NTOK - 1);
#pragma unroll
        for (int ct = 0; ct < 8; ++ct) {
            int d = n0 + ct * 16 + l16;
            u16x8 pk;
#pragma unroll
            for (int r = 0; r < 8; ++r) pk[r] = bfbits(acc[ct][r] * oscale);
            *(uint4*)(&Out[(size_t)b * DDIM * NTOK + (size_t)d * NTOK + tok]) =
                __builtin_bit_cast(uint4, pk);
        }
    }
}

// ---------------------------------------------------------------------------
// Attention: one WG = (batch b, 16 query rows). 256 threads = 8 waves.
// Dynamic LDS: S panel 16x2048 f32 (128KB) + Pbf panel 16x2048 bf16 (64KB).
// ---------------------------------------------------------------------------
__global__ __launch_bounds__(256) void attn_kernel(
    const unsigned short* __restrict__ Qbf,   // [B*N][D] bf16 (pre-scaled 1/sqrt(D))
    const unsigned short* __restrict__ Kbf,   // [B*N][D] bf16
    const unsigned short* __restrict__ VTbf,  // [B][D][N] bf16
    const float* __restrict__ imp,            // [N]
    float* __restrict__ out,                  // [B][N][D]
    float* __restrict__ attn,                 // [B][N][N]
    float* __restrict__ ent)                  // scalar
{
    extern __shared__ char smem_raw[];
    float* S            = (float*)smem_raw;                        // [16][2048] f32
    unsigned short* Pbf = (unsigned short*)(smem_raw + 16 * NTOK * 4); // [16][2048] bf16
    __shared__ unsigned short Qp[16 * DDIM];                       // 16KB q panel
    __shared__ float red[256];
    __shared__ float rowmax[16], rowsum[16];

    const int b    = blockIdx.y;
    const int q0   = blockIdx.x * 16;
    const int t    = threadIdx.x;
    const int wave = t >> 5, lane = t & 31;
    const int half = lane >> 4, l16 = lane & 15;

    // ---- load Q panel (16 x 512 bf16 = 1024 uint4) ----
    {
        const uint4* src = (const uint4*)(Qbf + (size_t)(b * NTOK + q0) * DDIM);
        uint4* dst = (uint4*)Qp;
        for (int j = t; j < 1024; j += 256) dst[j] = src[j];
    }
    __syncthreads();

    // ---- stage 1: S = Q @ K^T (Q pre-scaled). Wave w owns key cols [w*256,+256) ----
#pragma unroll
    for (int ctg = 0; ctg < 4; ++ctg) {
        f32x8 acc4[4];
        f32x8 zero = {};
#pragma unroll
        for (int i = 0; i < 4; ++i) acc4[i] = zero;

        for (int kt = 0; kt < 16; ++kt) {
            const int k0 = kt * 32;
            u16x16 au = load32B(&Qp[l16 * DDIM + k0 + half * 16]);
            u16x16 bu[4];
#pragma unroll
            for (int i = 0; i < 4; ++i) {
                int key = wave * 256 + (ctg * 4 + i) * 16 + l16;
                bu[i] = load32B(Kbf + (size_t)(b * NTOK + key) * DDIM + k0 + half * 16);
            }
#pragma unroll
            for (int i = 0; i < 4; ++i) acc4[i] = wmma_bb(au, bu[i], acc4[i]);
        }
#pragma unroll
        for (int i = 0; i < 4; ++i) {
            int n0 = wave * 256 + (ctg * 4 + i) * 16;
#pragma unroll
            for (int r = 0; r < 8; ++r)
                S[(r + 8 * half) * NTOK + n0 + l16] = acc4[i][r];
        }
    }
    __syncthreads();

    // ---- diagonal self-importance bias ----
    if (t < 16) {
        int qi = q0 + t;
        S[t * NTOK + qi] += fminf(__expf(imp[qi]), 3.0f);
    }
    __syncthreads();

    // ---- softmax + attn write + entropy + bf16 P panel ----
    const int row = t >> 4, chunk = t & 15;
    float* Srow = &S[row * NTOK];
    unsigned short* Prow = &Pbf[row * NTOK];
    float lmax = -3.4e38f;
    for (int i = 0; i < 128; ++i) lmax = fmaxf(lmax, Srow[chunk + 16 * i]);
    red[t] = lmax; __syncthreads();
    if (chunk == 0) {
        float m = red[t];
        for (int j = 1; j < 16; ++j) m = fmaxf(m, red[t + j]);
        rowmax[row] = m;
    }
    __syncthreads();
    const float rm = rowmax[row];
    float lsum = 0.f;
    for (int i = 0; i < 128; ++i) {
        int c = chunk + 16 * i;
        float e = __expf(Srow[c] - rm);
        Srow[c] = e; lsum += e;
    }
    red[t] = lsum; __syncthreads();
    if (chunk == 0) {
        float s = 0.f;
        for (int j = 0; j < 16; ++j) s += red[t + j];
        rowsum[row] = s;
    }
    __syncthreads();
    const float inv = 1.0f / rowsum[row];
    float lent = 0.f;
    float* arow = attn + ((size_t)b * NTOK + q0 + row) * NTOK;
    for (int i = 0; i < 128; ++i) {
        int c = chunk + 16 * i;
        float p = Srow[c] * inv;
        arow[c] = p;
        Prow[c] = bfbits(p);
        lent += p * __logf(p + 1e-10f);
    }
    red[t] = lent; __syncthreads();
    if (t == 0) {
        float s = 0.f;
        for (int j = 0; j < 256; ++j) s += red[j];
        atomicAdd(ent, -s * (1.0f / (float)(BATCH * NTOK)));
    }
    __syncthreads();

    // ---- stage 2: out = P @ V ; wave w owns out dims [w*64, w*64+64) ----
    f32x8 oacc[4];
    f32x8 zero = {};
#pragma unroll
    for (int i = 0; i < 4; ++i) oacc[i] = zero;

    for (int kt = 0; kt < 64; ++kt) {
        const int k0 = kt * 32;
        u16x16 au = load32B(&Pbf[l16 * NTOK + k0 + half * 16]);
        u16x16 bu[4];
#pragma unroll
        for (int c2 = 0; c2 < 4; ++c2) {
            int d = wave * 64 + c2 * 16 + l16;
            bu[c2] = load32B(VTbf + ((size_t)b * DDIM + d) * NTOK + k0 + half * 16);
        }
#pragma unroll
        for (int c2 = 0; c2 < 4; ++c2) oacc[c2] = wmma_bb(au, bu[c2], oacc[c2]);
    }
#pragma unroll
    for (int c2 = 0; c2 < 4; ++c2) {
        int d = wave * 64 + c2 * 16 + l16;
#pragma unroll
        for (int r = 0; r < 8; ++r) {
            int q = q0 + r + 8 * half;
            out[((size_t)b * NTOK + q) * DDIM + d] = oacc[c2][r];
        }
    }
}

__global__ void init_ent_kernel(float* e) { *e = 0.f; }

// ---------------------------------------------------------------------------
extern "C" void kernel_launch(void* const* d_in, const int* in_sizes, int n_in,
                              void* d_out, int out_size, void* d_ws, size_t ws_size,
                              hipStream_t stream) {
    const float* query = (const float*)d_in[0];
    const float* key   = (const float*)d_in[1];
    const float* value = (const float*)d_in[2];
    const float* Wq    = (const float*)d_in[3];
    const float* Wk    = (const float*)d_in[4];
    const float* Wv    = (const float*)d_in[5];
    const float* imp   = (const float*)d_in[6];

    float* out  = (float*)d_out;                               // [B][N][D]
    float* attn = out  + (size_t)BATCH * NTOK * DDIM;          // [B][N][N]
    float* ent  = attn + (size_t)BATCH * NTOK * NTOK;          // scalar

    unsigned short* Qbf  = (unsigned short*)d_ws;              // [B*N][D] bf16
    unsigned short* Kbf  = Qbf + (size_t)BATCH * NTOK * DDIM;
    unsigned short* VTbf = Kbf + (size_t)BATCH * NTOK * DDIM;  // [B][D][N] bf16

    const float qscale = 0.044194173824159216f;  // 1/sqrt(512), folded into Q

    dim3 pgrid((BATCH * NTOK) / 128, DDIM / 128);  // 128 x 4
    proj_kernel<<<pgrid, 256, 0, stream>>>(query, Wq, Qbf, qscale, 0);
    proj_kernel<<<pgrid, 256, 0, stream>>>(key,   Wk, Kbf, 1.0f, 0);
    proj_kernel<<<pgrid, 256, 0, stream>>>(value, Wv, VTbf, 1.0f, 1);

    init_ent_kernel<<<1, 1, 0, stream>>>(ent);

    const int smem_bytes = 16 * NTOK * 4 + 16 * NTOK * 2;  // 192KB S + Pbf
    (void)hipFuncSetAttribute((const void*)attn_kernel,
                              hipFuncAttributeMaxDynamicSharedMemorySize,
                              smem_bytes);
    dim3 agrid(NTOK / 16, BATCH);
    attn_kernel<<<agrid, 256, smem_bytes, stream>>>(Qbf, Kbf, VTbf, imp,
                                                    out, attn, ent);
}
